// TFN_24137716203561
// MI455X (gfx1250) — matrix operand order repaired and checked by reading, not compiled, not run on hardware
//
#include <hip/hip_runtime.h>

// ---------------------------------------------------------------------------
// Trilinear tensor-feature head for MI455X (gfx1250, wave32, WMMA + TDM).
//
//   out[b,o] = sum_{i,j,k} a[b,i]*p[b,j]*d[b,k] * W[(i*64+j)*64+k, o] + bias[o]
//
// One GEMM  out = A @ W  with A synthesized on the fly:
//   A[b, ij*64+k] = (a_i*p_j) * d_k        (bf16 fragments built in registers)
//   W streamed fp32 by the Tensor Data Mover (tensor_load_to_lds, TENSORcnt),
//   double-buffered in LDS; each of the 8 waves DMAs a 4-row slice per step
//   (1 instruction/wave/step vs 128 for per-thread async loads).
// Split-K across 128 blocks (x2 M-blocks), partials folded with f32 atomics.
// ---------------------------------------------------------------------------

typedef __attribute__((ext_vector_type(16))) __bf16 v16bf;
typedef __attribute__((ext_vector_type(8)))  float  v8f;
typedef __attribute__((ext_vector_type(4)))  float  f32x4;
typedef __attribute__((ext_vector_type(4)))  unsigned u32x4;
typedef __attribute__((ext_vector_type(8)))  unsigned u32x8;

#define BATCH   256
#define DIN     63
#define DP      64
#define NOUT    128
#define KK_TOT  (DP * DP * DP)           // 262144 flattened (i,j,k)
#define BLK_M   128
#define KSPLIT  128                      // grid.y: split-K factor
#define KK_BLK  (KK_TOT / KSPLIT)        // 2048 kk per block (= 32 ij values)
#define KSTEP   32                       // one bf16 WMMA K per step
#define NSTEPS  (KK_BLK / KSTEP)         // 64 main-loop iterations
#define WTILE_B (KSTEP * NOUT * 4)       // 16 KB fp32 W tile in LDS

// ---- TDM: DMA one wave's 4x128 fp32 slice of a W tile into LDS -------------
// D# group 0: [1:0]=count=1, [63:32]=lds_addr, [120:64]=global_addr,
//             [127:126]=type=2 ("image").
// D# group 1: data_size=4B, tensor_dim0=128, tensor_dim1=2^20 (no OOB),
//             tile_dim0=128, tile_dim1=4, tensor_dim0_stride=128.
__device__ __forceinline__ void tdm_load_w_slice(unsigned long long gaddr,
                                                 unsigned ldsaddr) {
    u32x4 g0;
    g0[0] = 1u;                                        // count=1 (valid user D#)
    g0[1] = ldsaddr;                                   // lds_addr (bytes)
    g0[2] = (unsigned)gaddr;                           // global_addr[31:0]
    g0[3] = (unsigned)(gaddr >> 32) | (2u << 30);      // addr[56:32] | type=2
    u32x8 g1;
    g1[0] = 2u << 16;                                  // data_size = 4 bytes
    g1[1] = (unsigned)NOUT << 16;                      // tensor_dim0 = 128
    g1[2] = 0u;                                        // dim0 hi / dim1 lo
    g1[3] = ((unsigned)NOUT << 16) | 16u;              // tile_dim0=128 | dim1 hi
    g1[4] = 4u;                                        // tile_dim1 = 4 rows
    g1[5] = (unsigned)NOUT;                            // tensor_dim0_stride=128
    g1[6] = 0u;
    g1[7] = 0u;
    asm volatile("tensor_load_to_lds %0, %1" :: "s"(g0), "s"(g1) : "memory");
}

// A fragment (16x32 MxK, bf16), built in registers from the d_pad panel.
// Lane L holds row M; halves 0..7 = K kb+hi*8+0..7, halves 8..15 = K+16.
__device__ __forceinline__ v16bf synth_a_frag(const float* dRow, float g,
                                              int kb, int hi) {
    const f32x4 d0 = *(const f32x4*)(dRow + kb + hi * 8);
    const f32x4 d1 = *(const f32x4*)(dRow + kb + hi * 8 + 4);
    const f32x4 d2 = *(const f32x4*)(dRow + kb + 16 + hi * 8);
    const f32x4 d3 = *(const f32x4*)(dRow + kb + 16 + hi * 8 + 4);
    union { v16bf v; __bf16 h[16]; } f;
#pragma unroll
    for (int t = 0; t < 4; ++t) {
        f.h[t]      = (__bf16)(g * d0[t]);
        f.h[4 + t]  = (__bf16)(g * d1[t]);
        f.h[8 + t]  = (__bf16)(g * d2[t]);
        f.h[12 + t] = (__bf16)(g * d3[t]);
    }
    return f.v;
}

// B fragment (32x16 KxN, bf16) from the fp32 W tile in LDS ([k][n] layout).
// Lane L holds col N = L&15; halves 0..7 = K hi*8+0..7, halves 8..15 = K+16.
__device__ __forceinline__ v16bf load_b_frag(const float* sWt, int col, int hi) {
    union { v16bf v; __bf16 h[16]; } f;
#pragma unroll
    for (int t = 0; t < 16; ++t)
        f.h[t] = (__bf16)sWt[(hi * 8 + (t & 7) + (t >> 3) * 16) * NOUT + col];
    return f.v;
}

__global__ void tfn_init_out(const float* __restrict__ bias,
                             float* __restrict__ out) {
    int idx = blockIdx.x * 256 + threadIdx.x;
    if (idx < BATCH * NOUT) out[idx] = bias[idx & (NOUT - 1)];
}

__global__ __launch_bounds__(256)
void tfn_trilinear_gemm(const float* __restrict__ x,
                        const float* __restrict__ prot,
                        const float* __restrict__ dyn,
                        const float* __restrict__ W,
                        float* __restrict__ out) {
    __shared__ float sDpad[BLK_M][DP];         // d with appended 1.0   (32 KB)
    __shared__ float sG[32][BLK_M];            // g[ij_local][row]      (16 KB)
    __shared__ float sW[2][KSTEP][NOUT];       // TDM W tiles, fp32     (32 KB)

    const int tid    = threadIdx.x;
    const int lane   = tid & 31;
    const int wave   = tid >> 5;
    const int b0     = blockIdx.x * BLK_M;
    const int ijBase = blockIdx.y * 32;
    const int kkBase = blockIdx.y * KK_BLK;

    // ---- stage d_pad (fp32): 128 rows x 64 cols, ones appended -------------
    {
        const int r  = tid >> 1;
        const int c0 = (tid & 1) * 32;
#pragma unroll 8
        for (int c = 0; c < 32; ++c) {
            const int cc = c0 + c;
            sDpad[r][cc] = (cc < DIN) ? dyn[(b0 + r) * DIN + cc] : 1.0f;
        }
    }
    // ---- precompute g[ij_local][row] = a_pad[row,i] * p_pad[row,j] ---------
    {
        const int v  = tid >> 3;
        const int rb = (tid & 7) * 16;
        const int ij = ijBase + v;
        const int i  = ij >> 6;
        const int j  = ij & 63;
#pragma unroll 4
        for (int rr = 0; rr < 16; ++rr) {
            const int r = rb + rr;
            const float av = (i < DIN) ? x[(b0 + r) * DIN + i] : 1.0f;
            const float pv = (j < DIN) ? prot[(b0 + r) * DIN + j] : 1.0f;
            sG[v][r] = av * pv;
        }
    }

    v8f acc[2][4] = {};                        // 32x64 wave tile

    const int wm   = wave >> 1;                // wave grid 4(M) x 2(N)
    const int wn   = wave & 1;
    const int hi   = lane >> 4;
    const int rowA = wm * 32 + (lane & 15);
    const int colB = wn * 64 + (lane & 15);

    const float* dRow0 = &sDpad[rowA][0];
    const float* dRow1 = &sDpad[rowA + 16][0];

    // ---- per-wave TDM slice addressing (all uniform -> SGPRs) --------------
    const unsigned waveU  = (unsigned)__builtin_amdgcn_readfirstlane(wave);
    const unsigned ldsW   = (unsigned)(uintptr_t)&sW[0][0][0];
    const unsigned sliceB = waveU * 4u * (unsigned)NOUT * 4u;   // 2 KB per wave
    const unsigned long long gW =
        (unsigned long long)(uintptr_t)W +
        (unsigned long long)kkBase * NOUT * 4ull + sliceB;

    // prologue: kick off tile 0 into buffer 0 (one TDM op per wave)
    tdm_load_w_slice(gW, ldsW + sliceB);

    for (int s = 0; s < NSTEPS; ++s) {
        const int ijl = s >> 1;                // ij constant per step
        const int kb  = (s & 1) * 32;          // k offset inside d_pad

        __builtin_amdgcn_s_wait_tensorcnt(0);  // this wave's slice of tile s
        __syncthreads();                       // all slices in, all reads done

        if (s + 1 < NSTEPS)                    // overlap DMA s+1 with compute
            tdm_load_w_slice(gW + (unsigned long long)(s + 1) * WTILE_B,
                             ldsW + (((s + 1) & 1) ? WTILE_B : 0u) + sliceB);
        if (s + 2 < NSTEPS)                    // pull s+2 toward L2
            __builtin_prefetch(W + (kkBase + (s + 2) * KSTEP) * NOUT +
                               lane * 4, 0, 1);

        // ---- A fragments: registers only, from sG / sDpad ------------------
        const float g0 = sG[ijl][rowA];
        const float g1 = sG[ijl][rowA + 16];
        v16bf af[2];
        af[0] = synth_a_frag(dRow0, g0, kb, hi);
        af[1] = synth_a_frag(dRow1, g1, kb, hi);

        // ---- B fragments from TDM-landed fp32 tile + WMMA ------------------
        const float* sWcur = &sW[s & 1][0][0];
        v16bf bfr[4];
#pragma unroll
        for (int ni = 0; ni < 4; ++ni)
            bfr[ni] = load_b_frag(sWcur, colB + ni * 16, hi);
#pragma unroll
        for (int mi = 0; mi < 2; ++mi)
#pragma unroll
            for (int ni = 0; ni < 4; ++ni)
                acc[mi][ni] = __builtin_amdgcn_wmma_f32_16x16x32_bf16(
                    false, af[mi], false, bfr[ni], (short)0, acc[mi][ni],
                    false, false);
    }

    // ---- split-K epilogue: fold partials with f32 atomics ------------------
    // C/D layout: VGPR v -> row v + (lane>=16 ? 8 : 0), col = lane&15.
    const int rOff = hi * 8;
    const int cOff = lane & 15;
#pragma unroll
    for (int mi = 0; mi < 2; ++mi) {
#pragma unroll
        for (int ni = 0; ni < 4; ++ni) {
            float* op = out + (b0 + wm * 32 + mi * 16 + rOff) * NOUT
                            + (wn * 64 + ni * 16 + cOff);
#pragma unroll
            for (int v = 0; v < 8; ++v)
                atomicAdd(op + v * NOUT, acc[mi][ni][v]);
        }
    }
}

extern "C" void kernel_launch(void* const* d_in, const int* in_sizes, int n_in,
                              void* d_out, int out_size, void* d_ws, size_t ws_size,
                              hipStream_t stream) {
    const float* x    = (const float*)d_in[0];
    const float* prot = (const float*)d_in[1];
    const float* dyn  = (const float*)d_in[2];
    const float* W    = (const float*)d_in[3];
    const float* bias = (const float*)d_in[4];
    float* out = (float*)d_out;

    tfn_init_out<<<(BATCH * NOUT + 255) / 256, 256, 0, stream>>>(bias, out);
    dim3 grid(BATCH / BLK_M, KSPLIT);          // (2, 128) workgroups
    tfn_trilinear_gemm<<<grid, 256, 0, stream>>>(x, prot, dyn, W, out);
}
